// DMPNNEncoder_27298812134157
// MI455X (gfx1250) — compile-verified, matrix operand order checked by reading, not run on hardware
//
#include <hip/hip_runtime.h>
#include <hip/hip_bf16.h>
#include <stdint.h>

#define HID     128
#define NODE_IN 133
#define EDGE_IN 14

typedef __attribute__((ext_vector_type(16))) __bf16 v16bf;
typedef __attribute__((ext_vector_type(8)))  float  v8f;

union ABFrag { unsigned u[8]; v16bf v; };
union CDFrag { float    f[8]; v8f   v; };

__device__ __forceinline__ unsigned short f2bf(float f) {
    unsigned u = __float_as_uint(f);
    u += 0x7fffu + ((u >> 16) & 1u);          // round-to-nearest-even
    return (unsigned short)(u >> 16);
}
__device__ __forceinline__ float bf2f(unsigned short h) {
    return __uint_as_float(((unsigned)h) << 16);
}
__device__ __forceinline__ void atomAdd(float* p, float v) {
    __hip_atomic_fetch_add(p, v, __ATOMIC_RELAXED, __HIP_MEMORY_SCOPE_AGENT);
}

// 16-bit WMMA operand fragment: lane holds row/col (lane&15), K-half koff=(lane>>4)*8,
// element pairs at K offsets {0,2,4,6, 16,18,20,22} relative to chunk base + koff.
__device__ __forceinline__ void load_frag(const unsigned short* p, ABFrag& fr) {
    fr.u[0] = *(const unsigned*)(p + 0);
    fr.u[1] = *(const unsigned*)(p + 2);
    fr.u[2] = *(const unsigned*)(p + 4);
    fr.u[3] = *(const unsigned*)(p + 6);
    fr.u[4] = *(const unsigned*)(p + 16);
    fr.u[5] = *(const unsigned*)(p + 18);
    fr.u[6] = *(const unsigned*)(p + 20);
    fr.u[7] = *(const unsigned*)(p + 22);
}

// Shared GEMM core: A-tile (64 x KPAD bf16) resident in LDS, B (weights, [n][KPAD] bf16
// in global) staged chunk-by-chunk. 4 waves, each computes 16 rows x 128 cols.
template<int KPAD, int LDA>
__device__ __forceinline__ void gemm_core(const unsigned short* __restrict__ ldsA,
                                          unsigned short* __restrict__ ldsB,
                                          const unsigned short* __restrict__ Wt,
                                          CDFrag acc[8]) {
    const int tid  = threadIdx.x;
    const int lane = tid & 31;
    const int wave = tid >> 5;
    const int mrow = lane & 15;
    const int koff = (lane >> 4) << 3;
    constexpr int NCHUNK = KPAD / 32;
    for (int kc = 0; kc < NCHUNK; ++kc) {
        for (int idx = tid; idx < 128 * 32; idx += 128) {      // stage B chunk [n][32]
            int n = idx >> 5, k = idx & 31;
            ldsB[n * 34 + k] = Wt[n * KPAD + kc * 32 + k];
        }
        __syncthreads();
        ABFrag a;
        load_frag(ldsA + (wave * 16 + mrow) * LDA + kc * 32 + koff, a);
#pragma unroll
        for (int ns = 0; ns < 8; ++ns) {
            ABFrag b;
            load_frag(ldsB + (ns * 16 + mrow) * 34 + koff, b);
            acc[ns].v = __builtin_amdgcn_wmma_f32_16x16x32_bf16(
                false, a.v, false, b.v, (short)0, acc[ns].v, false, false);
        }
        __syncthreads();
    }
}

// ---- GEMM1: h0 = relu(concat(x[src], edge_attr) @ W1 + b1); node_sum[dst] += h0 ----
__global__ __launch_bounds__(128) void k_gemm1(const float* __restrict__ x,
        const float* __restrict__ ea, const unsigned short* __restrict__ W1t,
        const float* __restrict__ b1, const int* __restrict__ src,
        const int* __restrict__ dst, unsigned short* __restrict__ h0,
        float* __restrict__ ns_out, int E) {
    constexpr int KPAD = 160, LDA = 162;
    __shared__ unsigned short ldsA[64 * LDA];
    __shared__ unsigned short ldsB[128 * 34];
    __shared__ int s_src[64], s_dst[64];
    const int tid = threadIdx.x;
    const int e0  = blockIdx.x * 64;
    if (tid < 64) {
        int e = e0 + tid;
        s_src[tid] = (e < E) ? src[e] : 0;
        s_dst[tid] = (e < E) ? dst[e] : 0;
    }
    __syncthreads();
    for (int idx = tid; idx < 64 * KPAD; idx += 128) {
        int r = idx / KPAD, c = idx - r * KPAD;
        int e = e0 + r;
        float v = 0.f;
        if (e < E) {
            if (c < NODE_IN)                 v = x[(long)s_src[r] * NODE_IN + c];
            else if (c < NODE_IN + EDGE_IN)  v = ea[(long)e * EDGE_IN + (c - NODE_IN)];
        }
        ldsA[r * LDA + c] = f2bf(v);
    }
    __syncthreads();
    CDFrag acc[8];
#pragma unroll
    for (int ns = 0; ns < 8; ++ns)
#pragma unroll
        for (int j = 0; j < 8; ++j) acc[ns].f[j] = 0.f;
    gemm_core<KPAD, LDA>(ldsA, ldsB, W1t, acc);
    const int lane = tid & 31, wave = tid >> 5;
    const int mrow = lane & 15, koff = (lane >> 4) << 3;
#pragma unroll
    for (int ns = 0; ns < 8; ++ns) {
        int n = ns * 16 + mrow;
        float bias = b1[n];
#pragma unroll
        for (int j = 0; j < 8; ++j) {
            int rloc = wave * 16 + koff + j;
            int e = e0 + rloc;
            if (e < E) {
                float v = fmaxf(acc[ns].f[j] + bias, 0.f);
                h0[(long)e * HID + n] = f2bf(v);
                atomAdd(&ns_out[(long)s_dst[rloc] * HID + n], v);
            }
        }
    }
}

// ---- GEMM2: hnext = relu(h0 + (ns_in[src] - hcur[rev]) @ W2 + b2); ns_out[dst] += hnext ----
__global__ __launch_bounds__(128) void k_gemm2(const float* __restrict__ ns_in,
        const unsigned short* __restrict__ hcur, const unsigned short* __restrict__ h0,
        const unsigned short* __restrict__ W2t, const float* __restrict__ b2,
        const int* __restrict__ src, const int* __restrict__ rev,
        const int* __restrict__ dst, unsigned short* __restrict__ hnext,
        float* __restrict__ ns_out, int E) {
    constexpr int KPAD = 128, LDA = 130;
    __shared__ unsigned short ldsA[64 * LDA];
    __shared__ unsigned short ldsB[128 * 34];
    __shared__ int s_src[64], s_rev[64], s_dst[64];
    const int tid = threadIdx.x;
    const int e0  = blockIdx.x * 64;
    if (tid < 64) {
        int e = e0 + tid;
        s_src[tid] = (e < E) ? src[e] : 0;
        s_rev[tid] = (e < E) ? rev[e] : 0;
        s_dst[tid] = (e < E) ? dst[e] : 0;
    }
    __syncthreads();
    for (int idx = tid; idx < 64 * 128; idx += 128) {
        int r = idx >> 7, c = idx & 127;
        float m = ns_in[(long)s_src[r] * HID + c] - bf2f(hcur[(long)s_rev[r] * HID + c]);
        ldsA[r * LDA + c] = f2bf(m);
    }
    __syncthreads();
    CDFrag acc[8];
#pragma unroll
    for (int ns = 0; ns < 8; ++ns)
#pragma unroll
        for (int j = 0; j < 8; ++j) acc[ns].f[j] = 0.f;
    gemm_core<KPAD, LDA>(ldsA, ldsB, W2t, acc);
    const int lane = tid & 31, wave = tid >> 5;
    const int mrow = lane & 15, koff = (lane >> 4) << 3;
#pragma unroll
    for (int ns = 0; ns < 8; ++ns) {
        int n = ns * 16 + mrow;
        float bias = b2[n];
#pragma unroll
        for (int j = 0; j < 8; ++j) {
            int rloc = wave * 16 + koff + j;
            int e = e0 + rloc;
            if (e < E) {
                float v = acc[ns].f[j] + bias + bf2f(h0[(long)e * HID + n]);
                v = fmaxf(v, 0.f);
                hnext[(long)e * HID + n] = f2bf(v);
                atomAdd(&ns_out[(long)s_dst[rloc] * HID + n], v);
            }
        }
    }
}

// ---- GEMM3: node_attr = relu(concat(x, node_sum) @ W3 + b3), fused pool-sum ----
__global__ __launch_bounds__(128) void k_gemm3(const float* __restrict__ x,
        const float* __restrict__ node_sum, const unsigned short* __restrict__ W3t,
        const float* __restrict__ b3, const int* __restrict__ batch,
        float* __restrict__ gsum, int N) {
    constexpr int KPAD = 288, LDA = 290;
    __shared__ unsigned short ldsA[64 * LDA];
    __shared__ unsigned short ldsB[128 * 34];
    __shared__ int s_batch[64];
    const int tid = threadIdx.x;
    const int v0  = blockIdx.x * 64;
    if (tid < 64) { int v = v0 + tid; s_batch[tid] = (v < N) ? batch[v] : 0; }
    __syncthreads();
    for (int idx = tid; idx < 64 * KPAD; idx += 128) {
        int r = idx / KPAD, c = idx - r * KPAD;
        int v = v0 + r;
        float val = 0.f;
        if (v < N) {
            if (c < NODE_IN)            val = x[(long)v * NODE_IN + c];
            else if (c < NODE_IN + HID) val = node_sum[(long)v * HID + (c - NODE_IN)];
        }
        ldsA[r * LDA + c] = f2bf(val);
    }
    __syncthreads();
    CDFrag acc[8];
#pragma unroll
    for (int ns = 0; ns < 8; ++ns)
#pragma unroll
        for (int j = 0; j < 8; ++j) acc[ns].f[j] = 0.f;
    gemm_core<KPAD, LDA>(ldsA, ldsB, W3t, acc);
    const int lane = tid & 31, wave = tid >> 5;
    const int mrow = lane & 15, koff = (lane >> 4) << 3;
#pragma unroll
    for (int ns = 0; ns < 8; ++ns) {
        int n = ns * 16 + mrow;
        float bias = b3[n];
#pragma unroll
        for (int j = 0; j < 8; ++j) {
            int rloc = wave * 16 + koff + j;
            int v = v0 + rloc;
            if (v < N) {
                float val = fmaxf(acc[ns].f[j] + bias, 0.f);
                atomAdd(&gsum[(long)s_batch[rloc] * HID + n], val);
            }
        }
    }
}

__global__ __launch_bounds__(256) void k_counts(const int* __restrict__ batch,
                                                float* __restrict__ gcount, int N) {
    int t = blockIdx.x * 256 + threadIdx.x;
    if (t < N) atomAdd(&gcount[batch[t]], 1.0f);
}

__global__ __launch_bounds__(256) void k_div(float* __restrict__ out,
                                             const float* __restrict__ gcount, int total) {
    int t = blockIdx.x * 256 + threadIdx.x;
    if (t < total) out[t] = out[t] / fmaxf(gcount[t >> 7], 1.0f);
}

// transpose + pad + bf16-convert weights: W[k][128] f32 -> Wt[n][KPAD] bf16
__global__ __launch_bounds__(256) void k_prepw(const float* __restrict__ W,
        unsigned short* __restrict__ Wt, int K, int KPAD) {
    int t = blockIdx.x * 256 + threadIdx.x;
    if (t < HID * KPAD) {
        int n = t / KPAD, k = t - n * KPAD;
        Wt[t] = (k < K) ? f2bf(W[(long)k * HID + n]) : (unsigned short)0;
    }
}

extern "C" void kernel_launch(void* const* d_in, const int* in_sizes, int n_in,
                              void* d_out, int out_size, void* d_ws, size_t ws_size,
                              hipStream_t stream) {
    (void)n_in; (void)ws_size;
    const float* x    = (const float*)d_in[0];
    const float* ea   = (const float*)d_in[1];
    const float* W1   = (const float*)d_in[2];
    const float* b1   = (const float*)d_in[3];
    const float* W2   = (const float*)d_in[4];
    const float* b2   = (const float*)d_in[5];
    const float* W3   = (const float*)d_in[6];
    const float* b3   = (const float*)d_in[7];
    const int* eidx   = (const int*)d_in[8];
    const int* rev    = (const int*)d_in[9];
    const int* batch  = (const int*)d_in[10];
    const int E = in_sizes[9];
    const int N = in_sizes[10];
    const int G = out_size / HID;
    const int* src = eidx;
    const int* dst = eidx + E;

    char* ws = (char*)d_ws;
    size_t off = 0;
    auto carve = [&](size_t bytes) -> void* {
        void* p = ws + off; off += (bytes + 255) & ~(size_t)255; return p;
    };
    unsigned short* h0  = (unsigned short*)carve((size_t)E * HID * 2);
    unsigned short* hA  = (unsigned short*)carve((size_t)E * HID * 2);
    unsigned short* hB  = (unsigned short*)carve((size_t)E * HID * 2);
    float* nsA          = (float*)carve((size_t)N * HID * 4);
    float* nsB          = (float*)carve((size_t)N * HID * 4);
    unsigned short* W1t = (unsigned short*)carve((size_t)HID * 160 * 2);
    unsigned short* W2t = (unsigned short*)carve((size_t)HID * 128 * 2);
    unsigned short* W3t = (unsigned short*)carve((size_t)HID * 288 * 2);
    float* gcount       = (float*)carve((size_t)G * 4);

    k_prepw<<<(HID * 160 + 255) / 256, 256, 0, stream>>>(W1, W1t, NODE_IN + EDGE_IN, 160);
    k_prepw<<<(HID * 128 + 255) / 256, 256, 0, stream>>>(W2, W2t, HID, 128);
    k_prepw<<<(HID * 288 + 255) / 256, 256, 0, stream>>>(W3, W3t, NODE_IN + HID, 288);

    const size_t nsBytes = (size_t)N * HID * 4;
    const int eblocks = (E + 63) / 64;

    // GEMM1 produces h0 and its segment-sum (into nsA)
    hipMemsetAsync(nsA, 0, nsBytes, stream);
    k_gemm1<<<eblocks, 128, 0, stream>>>(x, ea, W1t, b1, src, dst, h0, nsA, E);

    // three message-passing rounds, ping-ponging node_sum and h buffers
    unsigned short* cur = h0;
    unsigned short* hbufs[2] = {hA, hB};
    float* ns_in  = nsA;
    float* ns_out = nsB;
    for (int it = 0; it < 3; ++it) {
        unsigned short* nxt = hbufs[it & 1];
        hipMemsetAsync(ns_out, 0, nsBytes, stream);
        k_gemm2<<<eblocks, 128, 0, stream>>>(ns_in, cur, h0, W2t, b2, src, rev, dst,
                                             nxt, ns_out, E);
        cur = nxt;
        float* t = ns_in; ns_in = ns_out; ns_out = t;
    }
    // after the loop, ns_in holds segment_sum of the final h

    hipMemsetAsync(d_out, 0, (size_t)out_size * 4, stream);
    hipMemsetAsync(gcount, 0, (size_t)G * 4, stream);
    k_counts<<<(N + 255) / 256, 256, 0, stream>>>(batch, gcount, N);

    const int nblocks = (N + 63) / 64;
    k_gemm3<<<nblocks, 128, 0, stream>>>(x, ns_in, W3t, b3, batch, (float*)d_out, N);
    k_div<<<(out_size + 255) / 256, 256, 0, stream>>>((float*)d_out, gcount, out_size);
}
